// CodebookMapper_63290638074429
// MI455X (gfx1250) — matrix operand order repaired and checked by reading, not compiled, not run on hardware
//
#include <hip/hip_runtime.h>
#include <hip/hip_bf16.h>
#include <stdint.h>

typedef __bf16 bf16_t;
typedef __bf16 v8bf  __attribute__((ext_vector_type(8)));
typedef __bf16 v16bf __attribute__((ext_vector_type(16)));
typedef float  v8f   __attribute__((ext_vector_type(8)));
typedef uint32_t u32x4 __attribute__((ext_vector_type(4)));
typedef int    i32x4 __attribute__((ext_vector_type(4)));
typedef int    i32x8 __attribute__((ext_vector_type(8)));

#define N_ROWS 32768
#define K_CB   8192
#define D_DIM  512
#define TAU    0.07f

#define BM 32
#define BK 64
#define NIT (K_CB / BK)          // 128 iterations
#define THREADS 256

// ---------------- LDS layout (bytes). Row pads (+16B) make row strides
// 65 / 9 chunks of 16B -> co-prime with 16 -> conflict-free b128 frag loads.
#define Q_STRIDE   (D_DIM * 2 + 16)            // 1040
#define Q_OFF      0
#define Q_BYTES    (BM * Q_STRIDE)             // 33280
#define CBR_STRIDE (D_DIM * 2 + 16)            // 1040
#define CBR_BYTES  (BK * CBR_STRIDE)           // 66560
#define CBR_OFF(b) (Q_BYTES + (b) * CBR_BYTES)
#define CBT_OFF    (Q_BYTES + 2 * CBR_BYTES)   // 166400
#define CBT_STRIDE (BK * 2 + 16)               // 144
#define CBT_BYTES  (D_DIM * CBT_STRIDE)        // 73728
#define P_OFF      (CBT_OFF + CBT_BYTES)       // 240128
#define P_STRIDE   (BK * 2 + 16)               // 144
#define P_BYTES    (BM * P_STRIDE)             // 4608
#define STAT_OFF   (P_OFF + P_BYTES)           // 244736
#define SMEM_BYTES (STAT_OFF + 384 * 4)        // 246272  (< 320KB WGP LDS)

// ---------------- TDM: build a 2D D# (groups per CDNA5 ISA) and
// issue tensor_load_to_lds. Called from one wave only (EXEC-independent op).
__device__ __forceinline__ void tdm_load_2d(uint32_t lds_off, const void* gptr,
    uint32_t tensor_d0, uint32_t tensor_d1, uint32_t tile_d0, uint32_t tile_d1,
    uint64_t d0_stride_elems, uint32_t pad_interval_code, uint32_t pad_amount_code)
{
#if __has_builtin(__builtin_amdgcn_tensor_load_to_lds)
  uint64_t ga = (uint64_t)(uintptr_t)gptr;
  u32x4 g0;
  g0[0] = 1u;                                            // count=1, user mode
  g0[1] = lds_off;                                       // lds_addr
  g0[2] = (uint32_t)ga;                                  // global_addr[31:0]
  g0[3] = (uint32_t)((ga >> 32) & 0x1FFFFFFu) | (2u << 30); // addr[56:32] | type=2
  i32x8 g1;
  g1[0] = (int)((1u << 16)                                // data_size = 2B
              | (1u << 20)                                // pad_enable
              | (pad_interval_code << 22)
              | (pad_amount_code << 25));
  g1[1] = (int)((tensor_d0 & 0xFFFFu) << 16);             // tensor_dim0[15:0]
  g1[2] = (int)(((tensor_d0 >> 16) & 0xFFFFu) | ((tensor_d1 & 0xFFFFu) << 16));
  g1[3] = (int)(((tensor_d1 >> 16) & 0xFFFFu) | (tile_d0 << 16));
  g1[4] = (int)(tile_d1 & 0xFFFFu);                       // tile_dim1 (tile_dim2=0)
  g1[5] = (int)(uint32_t)(d0_stride_elems & 0xFFFFFFFFu); // dim0_stride[31:0]
  g1[6] = (int)(uint32_t)((d0_stride_elems >> 32) & 0xFFFFu);
  g1[7] = 0;
  i32x4 z4 = {0, 0, 0, 0};
#if defined(__clang_major__) && (__clang_major__ >= 23)
  i32x8 z8 = {0, 0, 0, 0, 0, 0, 0, 0};
  __builtin_amdgcn_tensor_load_to_lds(g0, g1, z4, z4, z8, 0);
#else
  __builtin_amdgcn_tensor_load_to_lds(g0, g1, z4, z4, 0);
#endif
#endif
}

// ---------------- WMMA fragment loaders (bf16, 16x16x32 shapes).
// A 16x32: lane: row=lane&15; lanes<16 hold k 0-7 & 16-23, lanes>=16 k 8-15 & 24-31.
__device__ __forceinline__ v16bf load_frag_a(const char* base, int stride, int lane) {
  const char* rp = base + (lane & 15) * stride + ((lane & 16) ? 16 : 0);
  v8bf lo = *(const v8bf*)rp;
  v8bf hi = *(const v8bf*)(rp + 32);
  return __builtin_shufflevector(lo, hi, 0,1,2,3,4,5,6,7,8,9,10,11,12,13,14,15);
}
// B 32x16: lane: col=lane&15 (= "row" of the LDS tile); lanes<16 hold k 0-7 & 16-23, lanes>=16 k 8-15 & 24-31.
__device__ __forceinline__ v16bf load_frag_b(const char* base, int stride, int lane) {
  const char* rp = base + (lane & 15) * stride + ((lane & 16) ? 16 : 0);
  v8bf lo = *(const v8bf*)rp;
  v8bf hi = *(const v8bf*)(rp + 32);
  return __builtin_shufflevector(lo, hi, 0,1,2,3,4,5,6,7,8,9,10,11,12,13,14,15);
}

// ---------------- one-time prep: codebook f32 -> bf16 row-major + bf16 d-major
__global__ void prep_codebook(const float* __restrict__ cb,
                              char* __restrict__ cb16, char* __restrict__ cbT16) {
  const int k  = blockIdx.x;            // 8192 rows
  const int d0 = threadIdx.x * 2;       // 256 threads * 2 = 512 cols
  float2 v = *(const float2*)(cb + (size_t)k * D_DIM + d0);
  bf16_t b0 = (bf16_t)v.x, b1 = (bf16_t)v.y;
  *(bf16_t*)(cb16 + ((size_t)k * D_DIM + d0) * 2)     = b0;
  *(bf16_t*)(cb16 + ((size_t)k * D_DIM + d0 + 1) * 2) = b1;
  *(bf16_t*)(cbT16 + ((size_t)d0 * K_CB + k) * 2)       = b0;
  *(bf16_t*)(cbT16 + ((size_t)(d0 + 1) * K_CB + k) * 2) = b1;
}

// ---------------- fused flash softmax-matmul kernel
__global__ void __launch_bounds__(THREADS, 1)
cbmap_fused(const float* __restrict__ x, const char* __restrict__ cb16,
            const char* __restrict__ cbT16, float* __restrict__ out) {
  extern __shared__ char smem[];
  const int tid  = threadIdx.x;
  const int lane = tid & 31;
  const int wave = tid >> 5;
  const int rbase = blockIdx.x * BM;

  // Kick off the first row-major codebook tile ASAP (overlaps Q preamble).
  if (wave == 0) {
    tdm_load_2d((uint32_t)(uintptr_t)(smem + CBR_OFF(0)), cb16,
                D_DIM, K_CB, D_DIM, BK, D_DIM, /*pad: every 256 dw*/7, /*4 dw*/3);
  }

  // ---- Q preamble: L2-normalize, fold 1/TAU, bf16 into LDS ----
  {
    const int qrow = tid >> 3;          // 32 rows, 8 threads per row
    const int dseg = (tid & 7) * 64;
    const float4* xp = (const float4*)(x + (size_t)(rbase + qrow) * D_DIM + dseg);
    float ss = 0.f;
#pragma unroll
    for (int i = 0; i < 16; ++i) {
      float4 v = xp[i];
      ss += v.x * v.x + v.y * v.y + v.z * v.z + v.w * v.w;
    }
    ss += __shfl_xor(ss, 1, 32);
    ss += __shfl_xor(ss, 2, 32);
    ss += __shfl_xor(ss, 4, 32);
    float scale = 1.0f / (fmaxf(sqrtf(ss), 1e-12f) * TAU);
    char* qp = smem + Q_OFF + qrow * Q_STRIDE + dseg * 2;
#pragma unroll
    for (int i = 0; i < 8; ++i) {
      float4 a = xp[2 * i], b = xp[2 * i + 1];
      v8bf pk;
      pk[0] = (bf16_t)(a.x * scale); pk[1] = (bf16_t)(a.y * scale);
      pk[2] = (bf16_t)(a.z * scale); pk[3] = (bf16_t)(a.w * scale);
      pk[4] = (bf16_t)(b.x * scale); pk[5] = (bf16_t)(b.y * scale);
      pk[6] = (bf16_t)(b.z * scale); pk[7] = (bf16_t)(b.w * scale);
      *(v8bf*)(qp + 16 * i) = pk;
    }
  }

  float* m_run   = (float*)(smem + STAT_OFF);
  float* l_run   = m_run + 32;
  float* m_cur   = m_run + 64;
  float* alpha_s = m_run + 96;
  float* pmax    = m_run + 128;   // [4][32]
  float* psum    = pmax + 128;    // [4][32]
  if (tid < 32) { m_run[tid] = -__builtin_inff(); l_run[tid] = 0.f; }

  v8f acc[2][4];
  v8f zero8 = {0.f, 0.f, 0.f, 0.f, 0.f, 0.f, 0.f, 0.f};
#pragma unroll
  for (int i = 0; i < 2; ++i)
#pragma unroll
    for (int j = 0; j < 4; ++j) acc[i][j] = zero8;

  __syncthreads();

  const int rs = wave >> 2;   // S row strip (0..1)
  const int ct = wave & 3;    // S col tile  (0..3)

  for (int it = 0; it < NIT; ++it) {
    const int has_next = (it + 1 < NIT);
    if (wave == 0) {
      // d-major tile for this iteration (single buffer, freed by trailing barrier)
      tdm_load_2d((uint32_t)(uintptr_t)(smem + CBT_OFF),
                  cbT16 + (size_t)it * BK * 2,
                  K_CB, D_DIM, BK, D_DIM, K_CB, /*pad: every 32 dw*/4, /*4 dw*/3);
      if (has_next) {  // prefetch next row-major tile into the other buffer
        tdm_load_2d((uint32_t)(uintptr_t)(smem + CBR_OFF((it + 1) & 1)),
                    cb16 + (size_t)(it + 1) * BK * D_DIM * 2,
                    D_DIM, K_CB, D_DIM, BK, D_DIM, 7, 3);
      }
      // ensure CBrow(it) (oldest outstanding) is complete; keep the rest in flight
      if (has_next) __builtin_amdgcn_s_wait_tensorcnt(2);
      else          __builtin_amdgcn_s_wait_tensorcnt(1);
    }
    __syncthreads();

    // ---- S phase: s(16x16) = Qstrip(16x512) x CBtile(512x16), 16 WMMAs ----
    const char* qbase = smem + Q_OFF + rs * 16 * Q_STRIDE;
    const char* kbase = smem + CBR_OFF(it & 1) + ct * 16 * CBR_STRIDE;
    v8f s = zero8;
#pragma unroll
    for (int kd = 0; kd < 16; ++kd) {
      v16bf a = load_frag_a(qbase + kd * 64, Q_STRIDE, lane);
      v16bf b = load_frag_b(kbase + kd * 64, CBR_STRIDE, lane);
      s = __builtin_amdgcn_wmma_f32_16x16x32_bf16(false, a, false, b,
                                                  (short)0, s, false, false);
    }

    // ---- per-row max of this tile (C layout: VGPR r = rows r / r+8) ----
    float red[8];
#pragma unroll
    for (int r = 0; r < 8; ++r) {
      float v = s[r];
      v = fmaxf(v, __shfl_xor(v, 1, 32));
      v = fmaxf(v, __shfl_xor(v, 2, 32));
      v = fmaxf(v, __shfl_xor(v, 4, 32));
      v = fmaxf(v, __shfl_xor(v, 8, 32));
      red[r] = v;
    }
    {
      float v = red[0];
#pragma unroll
      for (int r = 1; r < 8; ++r) if ((lane & 7) == r) v = red[r];
      if ((lane & 15) < 8)
        pmax[ct * 32 + rs * 16 + (lane & 7) + ((lane >> 4) << 3)] = v;
    }
    __syncthreads();
    if (tid < 32) {
      float mi = fmaxf(fmaxf(pmax[tid], pmax[32 + tid]),
                       fmaxf(pmax[64 + tid], pmax[96 + tid]));
      float mo = m_run[tid];
      float mc = fmaxf(mo, mi);
      m_cur[tid]   = mc;
      alpha_s[tid] = __expf(mo - mc);   // first iter: exp(-inf)=0
    }
    __syncthreads();

    // ---- P = exp(s - m_cur), row sums, bf16 P into LDS ----
    {
      float mrow[8];
#pragma unroll
      for (int r = 0; r < 8; ++r)
        mrow[r] = m_cur[rs * 16 + r + ((lane >> 4) << 3)];
      float p[8];
#pragma unroll
      for (int r = 0; r < 8; ++r) p[r] = __expf(s[r] - mrow[r]);
#pragma unroll
      for (int r = 0; r < 8; ++r) {
        float v = p[r];
        v += __shfl_xor(v, 1, 32);
        v += __shfl_xor(v, 2, 32);
        v += __shfl_xor(v, 4, 32);
        v += __shfl_xor(v, 8, 32);
        red[r] = v;
      }
      float v = red[0];
#pragma unroll
      for (int r = 1; r < 8; ++r) if ((lane & 7) == r) v = red[r];
      if ((lane & 15) < 8)
        psum[ct * 32 + rs * 16 + (lane & 7) + ((lane >> 4) << 3)] = v;
      char* pb = smem + P_OFF;
#pragma unroll
      for (int r = 0; r < 8; ++r) {
        int row = rs * 16 + r + ((lane >> 4) << 3);
        int col = ct * 16 + (lane & 15);
        *(bf16_t*)(pb + row * P_STRIDE + col * 2) = (bf16_t)p[r];
      }
    }
    __syncthreads();
    if (tid < 32) {
      float srow = psum[tid] + psum[32 + tid] + psum[64 + tid] + psum[96 + tid];
      l_run[tid] = l_run[tid] * alpha_s[tid] + srow;
      m_run[tid] = m_cur[tid];
    }
    if (wave == 0) {   // CBT(it) must be resident before the O phase
      if (has_next) __builtin_amdgcn_s_wait_tensorcnt(1);
      else          __builtin_amdgcn_s_wait_tensorcnt(0);
    }
    __syncthreads();

    // ---- O phase: rescale by alpha, acc += P(32xBK) x C(BKx64-per-wave) ----
    {
#pragma unroll
      for (int i = 0; i < 2; ++i) {
#pragma unroll
        for (int r = 0; r < 8; ++r) {
          float al = alpha_s[i * 16 + r + ((lane >> 4) << 3)];
#pragma unroll
          for (int j = 0; j < 4; ++j) acc[i][j][r] *= al;
        }
      }
      const char* pb = smem + P_OFF;
      const char* tb = smem + CBT_OFF + wave * 64 * CBT_STRIDE;
#pragma unroll
      for (int ks = 0; ks < 2; ++ks) {
        v16bf ap0 = load_frag_a(pb + 0 * 16 * P_STRIDE + ks * 64, P_STRIDE, lane);
        v16bf ap1 = load_frag_a(pb + 1 * 16 * P_STRIDE + ks * 64, P_STRIDE, lane);
#pragma unroll
        for (int j = 0; j < 4; ++j) {
          v16bf bc = load_frag_b(tb + j * 16 * CBT_STRIDE + ks * 64, CBT_STRIDE, lane);
          acc[0][j] = __builtin_amdgcn_wmma_f32_16x16x32_bf16(false, ap0, false, bc,
                          (short)0, acc[0][j], false, false);
          acc[1][j] = __builtin_amdgcn_wmma_f32_16x16x32_bf16(false, ap1, false, bc,
                          (short)0, acc[1][j], false, false);
        }
      }
    }
    __syncthreads();   // frees CBT + P for next iteration
  }

  // ---- epilogue: divide by softmax denominator, store f32 ----
#pragma unroll
  for (int i = 0; i < 2; ++i) {
#pragma unroll
    for (int r = 0; r < 8; ++r) {
      int row = i * 16 + r + ((lane >> 4) << 3);
      float inv = 1.0f / l_run[row];
      size_t ro = (size_t)(rbase + row) * D_DIM + wave * 64 + (lane & 15);
#pragma unroll
      for (int j = 0; j < 4; ++j) out[ro + j * 16] = acc[i][j][r] * inv;
    }
  }
}

extern "C" void kernel_launch(void* const* d_in, const int* in_sizes, int n_in,
                              void* d_out, int out_size, void* d_ws, size_t ws_size,
                              hipStream_t stream) {
  const float* x  = (const float*)d_in[0];
  const float* cb = (const float*)d_in[1];
  char* cb16  = (char*)d_ws;                                   // 8 MB row-major bf16
  char* cbT16 = (char*)d_ws + (size_t)K_CB * D_DIM * 2;        // 8 MB d-major bf16

  prep_codebook<<<K_CB, THREADS, 0, stream>>>(cb, cb16, cbT16);

  hipFuncSetAttribute(reinterpret_cast<const void*>(&cbmap_fused),
                      hipFuncAttributeMaxDynamicSharedMemorySize, SMEM_BYTES);
  cbmap_fused<<<N_ROWS / BM, THREADS, SMEM_BYTES, stream>>>(
      x, cb16, cbT16, (float*)d_out);
}
